// extra_fea_73641509257663
// MI455X (gfx1250) — matrix operand order repaired and checked by reading, not compiled, not run on hardware
//
#include <hip/hip_runtime.h>

// ---------------- problem constants ----------------
#define BB   8
#define CC   64
#define HH   256
#define WW   256
#define HWH  65536        // H*W
#define GRP  16           // groupnorm groups = C/4
#define EPS  1e-5f

typedef __attribute__((ext_vector_type(16))) __bf16        v16bf;
typedef __attribute__((ext_vector_type(8)))  float         v8f;
typedef __attribute__((ext_vector_type(8)))  unsigned int  v8u;
typedef __attribute__((ext_vector_type(4)))  float         v4f;
typedef __attribute__((ext_vector_type(2)))  float         v2f;

// ---------------- helpers ----------------
__device__ inline unsigned short f2bfbits(float f) {
  unsigned u = __builtin_bit_cast(unsigned, f);
  u += 0x7fffu + ((u >> 16) & 1u);           // round-to-nearest-even
  return (unsigned short)(u >> 16);
}
__device__ inline unsigned fenc(float f) {   // order-preserving float->uint
  unsigned u = __builtin_bit_cast(unsigned, f);
  return (u & 0x80000000u) ? ~u : (u | 0x80000000u);
}
__device__ inline float fdec(unsigned u) {
  unsigned v = (u & 0x80000000u) ? (u & 0x7fffffffu) : ~u;
  return __builtin_bit_cast(float, v);
}
// fast sigmoid: v_exp_f32 + v_rcp_f32 (avoids IEEE div expansion + exec branch)
__device__ inline float sigm(float v) {
  return __builtin_amdgcn_rcpf(1.f + __expf(-v));
}

// ---------------- K_w : weights -> bf16, pre-swizzled to WMMA A-fragment order
// layout: frag[(t*2+h)*32 + lane][e] ; per-lane 16 bf16 = 32 contiguous bytes.
__global__ void kw_swz(const float* __restrict__ gw, const float* __restrict__ dw,
                       unsigned short* __restrict__ gsw, unsigned short* __restrict__ dsw) {
  const int slot = threadIdx.x;              // 256 = 4t * 2h * 32lane
  const int t = slot >> 6, h = (slot >> 5) & 1, lane = slot & 31;
  const int m = lane & 15, khi = lane >> 4;
  for (int e = 0; e < 16; ++e) {
    int K   = e + 16 * khi;                              // contiguous k order: k = 16*half + e
    int src = (t * 16 + m) * 64 + h * 32 + K;
    int dst = ((t * 2 + h) * 32 + lane) * 16 + e;
    gsw[dst] = f2bfbits(gw[src]);
    dsw[dst] = f2bfbits(dw[src]);
  }
}

// ---------------- K1 : fused pooling (one b128 pass over x) ----------------
__global__ void k1_pool(const float* __restrict__ x, float* __restrict__ smean,
                        float* __restrict__ smax, float* __restrict__ avgsum,
                        unsigned* __restrict__ maxbits) {
  __shared__ float    csum[CC];
  __shared__ unsigned cmax[CC];
  const int b = blockIdx.y, tid = threadIdx.x, lane = tid & 31;
  if (tid < CC) { csum[tid] = 0.f; cmax[tid] = 0u; }
  __syncthreads();
  const int p = (blockIdx.x * 256 + tid) * 4;            // 4 px per thread
  const size_t base = ((size_t)b * CC) << 16;
  v4f ps = {0.f, 0.f, 0.f, 0.f};
  v4f pm = {-3.4e38f, -3.4e38f, -3.4e38f, -3.4e38f};
  for (int c = 0; c < CC; ++c) {
    v4f v = *(const v4f*)&x[base + ((size_t)c << 16) + p];
    for (int j = 0; j < 4; ++j) { ps[j] += v[j]; pm[j] = fmaxf(pm[j], v[j]); }
    float s = v[0] + v[1] + v[2] + v[3];
    float mx = fmaxf(fmaxf(v[0], v[1]), fmaxf(v[2], v[3]));
    unsigned e = fenc(mx);
    for (int m = 1; m < 32; m <<= 1) {
      s += __shfl_xor(s, m, 32);
      unsigned o = __shfl_xor(e, m, 32);
      e = (o > e) ? o : e;
    }
    if (lane == 0) { atomicAdd(&csum[c], s); atomicMax(&cmax[c], e); }
  }
  v4f mn; for (int j = 0; j < 4; ++j) mn[j] = ps[j] * (1.f / 64.f);
  *(v4f*)&smean[(size_t)b * HWH + p] = mn;
  *(v4f*)&smax [(size_t)b * HWH + p] = pm;
  __syncthreads();
  if (tid < CC) {
    atomicAdd(&avgsum[b * CC + tid], csum[tid]);
    atomicMax(&maxbits[b * CC + tid], cmax[tid]);
  }
}

// ---------------- K2 : channel-attention MLP ----------------
__global__ void k2_ca(const float* __restrict__ avgsum, const unsigned* __restrict__ maxbits,
                      const float* __restrict__ w1, const float* __restrict__ w2,
                      float* __restrict__ ca) {
  __shared__ float va[CC], vm[CC], ha[4], hm[4];
  const int b = blockIdx.x, c = threadIdx.x;
  va[c] = avgsum[b * CC + c] * (1.f / (float)HWH);
  vm[c] = fdec(maxbits[b * CC + c]);
  __syncthreads();
  if (c < 4) {
    float sa_ = 0.f, sm_ = 0.f;
    for (int i = 0; i < CC; ++i) { sa_ += w1[c * CC + i] * va[i]; sm_ += w1[c * CC + i] * vm[i]; }
    ha[c] = fmaxf(sa_, 0.f); hm[c] = fmaxf(sm_, 0.f);
  }
  __syncthreads();
  float o = 0.f;
  for (int k = 0; k < 4; ++k) o += w2[c * 4 + k] * (ha[k] + hm[k]);
  ca[b * CC + c] = sigm(o);
}

// ---------------- K3 : spatial attention 7x7 conv ----------------
__global__ void k3_sa(const float* __restrict__ smean, const float* __restrict__ smax,
                      const float* __restrict__ w, float* __restrict__ sa) {
  const int b = blockIdx.y;
  const int p = blockIdx.x * 256 + threadIdx.x;
  const int y0 = p >> 8, x0 = p & 255;
  const float* m0 = smean + (size_t)b * HWH;
  const float* m1 = smax  + (size_t)b * HWH;
  float acc = 0.f;
  for (int ky = 0; ky < 7; ++ky) {
    int yy = y0 + ky - 3;
    if (yy < 0 || yy >= HH) continue;
    for (int kx = 0; kx < 7; ++kx) {
      int xx = x0 + kx - 3;
      if (xx < 0 || xx >= WW) continue;
      int a = yy * WW + xx;
      acc += w[ky * 7 + kx] * m0[a] + w[49 + ky * 7 + kx] * m1[a];
    }
  }
  sa[(size_t)b * HWH + p] = sigm(acc);
}

// ---------------- K4 : fused gate-GEMM + mask + dsc-GEMM (WMMA) ----------------
// per wave: one 16-pixel x 64-channel tile.
// B fragments live in LDS in exact WMMA register order -> one 32B read each.
// NOTE: weight/output pointers intentionally NOT __restrict__ so the compiler
// keeps the cheap L2-hot b128 fragment loads inside the loop (no spill-hoist).
__global__ void k4_gemm(const float* __restrict__ x, const float* __restrict__ sab,
                        const float* __restrict__ ca, const unsigned short* gsw,
                        const unsigned short* dsw, const float* __restrict__ dscb,
                        float* dbuf, float* __restrict__ gnsum, float* __restrict__ gnsq) {
  __shared__ v4f xs4[8 * 256];   // 8 waves * 1024 f32 : xT[n][c]
  __shared__ v8u bstg[8 * 64];   // 8 waves * [2 ksteps][32 lanes] * 32B bf16 frags
  __shared__ float gs[GRP], gq[GRP];
  const int b = blockIdx.y, tid = threadIdx.x;
  const int wave = tid >> 5, lane = tid & 31;
  const int khi = lane >> 4, nn = lane & 15;
  float* xw = (float*)xs4 + wave * 1024;
  unsigned short* bw = (unsigned short*)bstg + wave * 1024;
  const v8u* bwv = bstg + wave * 64;
  const v8u* g8 = (const v8u*)gsw;
  const v8u* d8 = (const v8u*)dsw;
  if (tid < GRP) { gs[tid] = 0.f; gq[tid] = 0.f; }
  __syncthreads();

#pragma unroll 1
  for (int it = 0; it < 4; ++it) {                       // 4096 tiles / 1024 waves
    const int tile = blockIdx.x * 8 + wave + it * 1024;
    const int p0 = tile * 16;
    const int row = p0 >> 8, col0 = p0 & 255;
    const float* xb = x + (((size_t)b * CC) << 16) + p0;
    const float sav = sab[(size_t)b * HWH + p0 + nn];

    // stage xT (f32) and fuse = x*(sa+ca) (bf16 pairs, fragment order) into LDS
#pragma unroll
    for (int i = 0; i < 16; ++i) {
      const int c0 = 4 * i + 2 * khi;                    // even channel pair
      float xv0 = xb[((size_t)c0 << 16) + nn];
      float xv1 = xb[((size_t)(c0 + 1) << 16) + nn];
      v2f xp = {xv0, xv1};
      *(v2f*)&xw[nn * 64 + c0] = xp;
      float f0 = xv0 * (sav + ca[b * CC + c0]);
      float f1 = xv1 * (sav + ca[b * CC + c0 + 1]);
      unsigned pk = (unsigned)f2bfbits(f0) | ((unsigned)f2bfbits(f1) << 16);
      const int h = c0 >> 5, li = ((c0 >> 4) & 1) * 16 + nn;
      *(unsigned*)&bw[h * 512 + li * 16 + (c0 & 15)] = pk;
    }
    __syncthreads();

    // ---- GEMM 1: guide = sigmoid(gate_w @ fuse) ----
    v8f acc[4];
    for (int t = 0; t < 4; ++t) for (int r = 0; r < 8; ++r) acc[t][r] = 0.f;
    for (int h = 0; h < 2; ++h) {
      v16bf bfr = __builtin_bit_cast(v16bf, bwv[h * 32 + lane]);
      for (int t = 0; t < 4; ++t) {
        v16bf afr = __builtin_bit_cast(v16bf, g8[(t * 2 + h) * 32 + lane]);
        acc[t] = __builtin_amdgcn_wmma_f32_16x16x32_bf16(
            false, afr, false, bfr, (short)0, acc[t], false, false);
      }
    }

    // sigmoid -> xg = x*guide, masked (bilinear last row/col) -> LDS bf16 frags
    const bool lastRow = (row == HH - 1);
    const bool lastCol = (col0 + nn == WW - 1);
    const float msk = (lastRow || lastCol) ? 0.f : 1.f;
    for (int t = 0; t < 4; ++t) {
      const int off = nn * 64 + t * 16 + 8 * khi;
      v4f xa = *(const v4f*)&xw[off];
      v4f xc = *(const v4f*)&xw[off + 4];
      float xr[8] = {xa[0], xa[1], xa[2], xa[3], xc[0], xc[1], xc[2], xc[3]};
      for (int r = 0; r < 8; r += 2) {
        float v0 = msk * xr[r]     * sigm(acc[t][r]);
        float v1 = msk * xr[r + 1] * sigm(acc[t][r + 1]);
        unsigned pk = (unsigned)f2bfbits(v0) | ((unsigned)f2bfbits(v1) << 16);
        *(unsigned*)&bw[(t >> 1) * 512 + ((t & 1) * 16 + nn) * 16 + r + 8 * khi] = pk;
      }
    }
    __syncthreads();

    // ---- GEMM 2: d = dsc_w @ xg + bias ----
    for (int t = 0; t < 4; ++t) for (int r = 0; r < 8; ++r) acc[t][r] = 0.f;
    for (int h = 0; h < 2; ++h) {
      v16bf bfr = __builtin_bit_cast(v16bf, bwv[h * 32 + lane]);
      for (int t = 0; t < 4; ++t) {
        v16bf afr = __builtin_bit_cast(v16bf, d8[(t * 2 + h) * 32 + lane]);
        acc[t] = __builtin_amdgcn_wmma_f32_16x16x32_bf16(
            false, afr, false, bfr, (short)0, acc[t], false, false);
      }
    }

    // epilogue: store d, GN partial sums (shuffle over the 16 pixel lanes)
    for (int t = 0; t < 4; ++t)
      for (int r = 0; r < 8; ++r) {
        const int c = t * 16 + r + 8 * khi;
        float dv = acc[t][r] + dscb[c];
        dbuf[(((size_t)(b * CC + c)) << 16) + p0 + nn] = dv;
        float s = dv, q = dv * dv;
        for (int m = 1; m < 16; m <<= 1) { s += __shfl_xor(s, m, 32); q += __shfl_xor(q, m, 32); }
        if (nn == 0) { atomicAdd(&gs[c >> 2], s); atomicAdd(&gq[c >> 2], q); }
      }
    __syncthreads();
  }
  if (tid < GRP) {
    atomicAdd(&gnsum[b * GRP + tid], gs[tid]);
    atomicAdd(&gnsq [b * GRP + tid], gq[tid]);
  }
}

// ---------------- K5 : finalize GN stats ----------------
__global__ void k5_gnfin(const float* __restrict__ gnsum, const float* __restrict__ gnsq,
                         float* __restrict__ gnm, float* __restrict__ gnr) {
  int i = threadIdx.x;                       // 128 = B*GRP
  const float inv = 1.f / (4.f * (float)HWH);
  float mu = gnsum[i] * inv;
  float var = gnsq[i] * inv - mu * mu;
  gnm[i] = mu;
  gnr[i] = rsqrtf(var + EPS);
}

// ---------------- K6 : BN stats of y = relu(gn(d)) + x (b128) ----------------
__global__ void k6_bnstats(const float* __restrict__ d, const float* __restrict__ x,
                           const float* __restrict__ gnm, const float* __restrict__ gnr,
                           const float* __restrict__ gg, const float* __restrict__ gbeta,
                           float* __restrict__ bnsum, float* __restrict__ bnsq) {
  __shared__ float rs[256], rq[256];
  const int c = blockIdx.x, tid = threadIdx.x;
  const int b = blockIdx.y >> 3;                         // 8 blocks per (b) slice
  const int g = c >> 2;
  const size_t pbase = (size_t)(blockIdx.y & 7) * 8192;  // pixels, 8192 per block
  const float gr = gnr[b * GRP + g];
  const float sA = gr * gg[c];
  const float sB = gbeta[c] - gnm[b * GRP + g] * sA;
  const size_t rowb = ((size_t)(b * CC + c)) << 16;
  float s = 0.f, q = 0.f;
  for (int i = tid; i < 2048; i += 256) {                // 2048 float4 per block
    size_t a = rowb + pbase + (size_t)i * 4;
    v4f dv = *(const v4f*)&d[a];
    v4f xv = *(const v4f*)&x[a];
    for (int j = 0; j < 4; ++j) {
      float y = fmaxf(dv[j] * sA + sB, 0.f) + xv[j];
      s += y; q += y * y;
    }
  }
  rs[tid] = s; rq[tid] = q;
  __syncthreads();
  for (int st = 128; st > 0; st >>= 1) {
    if (tid < st) { rs[tid] += rs[tid + st]; rq[tid] += rq[tid + st]; }
    __syncthreads();
  }
  if (tid == 0) { atomicAdd(&bnsum[c], rs[0]); atomicAdd(&bnsq[c], rq[0]); }
}

// ---------------- K6b : finalize BN stats ----------------
__global__ void k6b_bnfin(const float* __restrict__ bnsum, const float* __restrict__ bnsq,
                          float* __restrict__ bnm, float* __restrict__ bnr) {
  int c = threadIdx.x;
  const float inv = 1.f / ((float)BB * (float)HWH);
  float mu = bnsum[c] * inv;
  float var = bnsq[c] * inv - mu * mu;
  bnm[c] = mu;
  bnr[c] = rsqrtf(var + EPS);
}

// ---------------- K7 : final GN+res+BN+PReLU+out-conv (b128) ----------------
__global__ void k7_out(const float* __restrict__ d, const float* __restrict__ x,
                       const float* __restrict__ gnm, const float* __restrict__ gnr,
                       const float* __restrict__ gg, const float* __restrict__ gbeta,
                       const float* __restrict__ bnm, const float* __restrict__ bnr,
                       const float* __restrict__ bg, const float* __restrict__ bbeta,
                       const float* __restrict__ prelu, const float* __restrict__ ow,
                       const float* __restrict__ ob, float* __restrict__ out) {
  __shared__ float s1[CC], s0[CC], t1[CC], t0[CC], wv[CC];
  const int b = blockIdx.y, tid = threadIdx.x;
  if (tid < CC) {
    int c = tid, g = c >> 2;
    float gr = gnr[b * GRP + g];
    float A = gr * gg[c];
    s1[c] = A; s0[c] = gbeta[c] - gnm[b * GRP + g] * A;
    float Bc = bnr[c] * bg[c];
    t1[c] = Bc; t0[c] = bbeta[c] - bnm[c] * Bc;
    wv[c] = ow[c];
  }
  __syncthreads();
  const int p = (blockIdx.x * 256 + tid) * 4;            // 4 px per thread
  const size_t base = ((size_t)b * CC) << 16;
  const float aP = prelu[0];
  v4f acc = {0.f, 0.f, 0.f, 0.f};
  for (int c = 0; c < CC; ++c) {
    size_t a = base + ((size_t)c << 16) + p;
    if (c < CC - 1) {                                    // gfx1250 global_prefetch_b8
      __builtin_prefetch(&d[a + HWH], 0, 1);
      __builtin_prefetch(&x[a + HWH], 0, 1);
    }
    v4f dv = *(const v4f*)&d[a];
    v4f xv = *(const v4f*)&x[a];
    for (int j = 0; j < 4; ++j) {
      float y = fmaxf(dv[j] * s1[c] + s0[c], 0.f) + xv[j];
      float z = y * t1[c] + t0[c];
      z = (z >= 0.f) ? z : aP * z;
      acc[j] += wv[c] * z;
    }
  }
  v4f res;
  for (int j = 0; j < 4; ++j) res[j] = acc[j] + ob[0];
  *(v4f*)&out[(size_t)b * HWH + p] = res;
}

// ---------------- host launch ----------------
extern "C" void kernel_launch(void* const* d_in, const int* in_sizes, int n_in,
                              void* d_out, int out_size, void* d_ws, size_t ws_size,
                              hipStream_t stream) {
  (void)in_sizes; (void)n_in; (void)out_size; (void)ws_size;
  const float* x       = (const float*)d_in[0];
  const float* ca_w1   = (const float*)d_in[1];
  const float* ca_w2   = (const float*)d_in[2];
  const float* sa_w    = (const float*)d_in[3];
  const float* gate_w  = (const float*)d_in[4];
  // d_in[5..8]: offset branch -> dead in forward math, skipped
  const float* dsc_w   = (const float*)d_in[9];
  const float* dsc_b   = (const float*)d_in[10];
  const float* gn_g    = (const float*)d_in[11];
  const float* gn_b    = (const float*)d_in[12];
  const float* bn_g    = (const float*)d_in[13];
  const float* bn_b    = (const float*)d_in[14];
  const float* prelu_a = (const float*)d_in[15];
  const float* out_w   = (const float*)d_in[16];
  const float* out_b   = (const float*)d_in[17];
  float* out = (float*)d_out;

  // workspace carve-up (256B aligned)
  size_t off = 0;
  auto carve = [&](size_t bytes) {
    void* p = (char*)d_ws + off;
    off += (bytes + 255) & ~(size_t)255;
    return p;
  };
  float*          dbuf   = (float*)carve((size_t)BB * CC * HWH * 4);   // 134 MB
  float*          smean  = (float*)carve((size_t)BB * HWH * 4);
  float*          smax   = (float*)carve((size_t)BB * HWH * 4);
  float*          sab    = (float*)carve((size_t)BB * HWH * 4);
  size_t accStart = off;                     // zero-init zone begins here
  float*          avgsum = (float*)carve(BB * CC * 4);
  unsigned*       maxbit = (unsigned*)carve(BB * CC * 4);
  float*          cab    = (float*)carve(BB * CC * 4);
  unsigned short* gsw    = (unsigned short*)carve(64 * 64 * 2);
  unsigned short* dsw    = (unsigned short*)carve(64 * 64 * 2);
  float*          gnsum  = (float*)carve(BB * GRP * 4);
  float*          gnsq   = (float*)carve(BB * GRP * 4);
  float*          gnm    = (float*)carve(BB * GRP * 4);
  float*          gnr    = (float*)carve(BB * GRP * 4);
  float*          bnsum  = (float*)carve(CC * 4);
  float*          bnsq   = (float*)carve(CC * 4);
  float*          bnm    = (float*)carve(CC * 4);
  float*          bnr    = (float*)carve(CC * 4);
  size_t accBytes = off - accStart;

  hipMemsetAsync((char*)d_ws + accStart, 0, accBytes, stream);

  kw_swz<<<1, 256, 0, stream>>>(gate_w, dsc_w, gsw, dsw);
  k1_pool<<<dim3(HWH / 1024, BB), 256, 0, stream>>>(x, smean, smax, avgsum, maxbit);
  k2_ca<<<BB, CC, 0, stream>>>(avgsum, maxbit, ca_w1, ca_w2, cab);
  k3_sa<<<dim3(HWH / 256, BB), 256, 0, stream>>>(smean, smax, sa_w, sab);
  k4_gemm<<<dim3(128, BB), 256, 0, stream>>>(x, sab, cab, gsw, dsw, dsc_b,
                                             dbuf, gnsum, gnsq);
  k5_gnfin<<<1, BB * GRP, 0, stream>>>(gnsum, gnsq, gnm, gnr);
  k6_bnstats<<<dim3(CC, 64), 256, 0, stream>>>(dbuf, x, gnm, gnr, gn_g, gn_b,
                                               bnsum, bnsq);
  k6b_bnfin<<<1, CC, 0, stream>>>(bnsum, bnsq, bnm, bnr);
  k7_out<<<dim3(HWH / 1024, BB), 256, 0, stream>>>(dbuf, x, gnm, gnr, gn_g, gn_b,
                                                   bnm, bnr, bn_g, bn_b, prelu_a,
                                                   out_w, out_b, out);
}
